// ProductVQLayer_7902739824790
// MI455X (gfx1250) — compile-verified, matrix operand order checked by reading, not run on hardware
//
#include <hip/hip_runtime.h>

typedef __attribute__((ext_vector_type(16))) _Float16 v16h;
typedef __attribute__((ext_vector_type(8)))  float    v8f;
typedef __attribute__((ext_vector_type(4)))  int      v4i;

#define B_TOTAL        131072
#define KCODES         2048
#define NSLICE         5
#define FDIM           36
#define ROWS_PER_BLOCK 256                    // 8 waves x 2 row-tiles x 16 rows
#define TILES_K        (KCODES / 16)          // 128 K-tiles of 16 codes
#define TILES_PER_CHUNK 32                    // 32 KB LDS chunk of B fragments
#define NCHUNK         (TILES_K / TILES_PER_CHUNK)   // 4 chunks/slice, 20 total
#define FRAG_HALFS     (32 * 16)              // one 16x16 tile fragment: 32 lanes x 16 halfs
#define SLICE_FRAG_HALFS (TILES_K * FRAG_HALFS)
#define CHUNK_FRAG_HALFS (TILES_PER_CHUNK * FRAG_HALFS)   // 16384 halfs = 32 KB
#define CHUNK_BYTES    (CHUNK_FRAG_HALFS * 2)
#define B128_PER_THREAD (CHUNK_BYTES / 16 / 256)          // 8

// ---- CDNA5 async global->LDS path (guarded: falls back to plain loads) ----
#if defined(__HIP_DEVICE_COMPILE__) && __has_builtin(__builtin_amdgcn_global_load_async_to_lds_b128) && __has_builtin(__builtin_amdgcn_s_wait_asynccnt)
#define HAVE_ASYNC 1
#else
#define HAVE_ASYNC 0
#endif

#if HAVE_ASYNC
// signature (from hipcc diagnostic): (as1 v4i*, as3 v4i*, imm offset, imm cpol)
#define ASYNC_COPY16(g, l)                                                    \
    __builtin_amdgcn_global_load_async_to_lds_b128(                           \
        (__attribute__((address_space(1))) v4i*)(g),                          \
        (__attribute__((address_space(3))) v4i*)(l), 0, 0)
#define ASYNC_WAIT(n) __builtin_amdgcn_s_wait_asynccnt(n)
#else
#define ASYNC_COPY16(g, l) (*(float4*)(l) = *(const float4*)(g))
#define ASYNC_WAIT(n)
#endif

// ---------------------------------------------------------------------------
// Pack codebooks into WMMA B-operand fragments (f16), fused score layout:
//   b[k] = -2048*cb[n][k] (k<d) ; b[d] = 1024*||cb[n]||^2 ; else 0.
// Scores = 1024 * (||cb||^2 - 2 f.cb)  -> argmin invariant, f16-safe range.
// B 32x16 f16 layout (mirrors ISA 16-bit A 16x32, N in place of M):
//   lanes 0-15 : N = lane,     halfs 0-7 = K 0..7,   halfs 8-15 = K 16..23
//   lanes16-31 : N = lane-16,  halfs 0-7 = K 8..15,  halfs 8-15 = K 24..31
// ---------------------------------------------------------------------------
__global__ void vq_pack(const float* __restrict__ cb0, const float* __restrict__ cb1,
                        const float* __restrict__ cb2, const float* __restrict__ cb3,
                        const float* __restrict__ cb4,
                        _Float16* __restrict__ bpack, float* __restrict__ lossAcc) {
    const int gid = blockIdx.x * blockDim.x + threadIdx.x;
    if (gid < NSLICE) lossAcc[gid] = 0.0f;          // re-zero every launch
    if (gid >= NSLICE * KCODES) return;

    const int s = gid / KCODES;
    const int n = gid % KCODES;
    const float* cbs[NSLICE] = {cb0, cb1, cb2, cb3, cb4};
    const int DIM[NSLICE] = {10, 6, 6, 9, 5};

    const int d = DIM[s];
    const float* row = cbs[s] + n * d;
    float norm = 0.0f;
    for (int k = 0; k < d; ++k) { float c = row[k]; norm += c * c; }

    const int t  = n >> 4;
    const int nn = n & 15;
    _Float16* frag = bpack + (size_t)(s * TILES_K + t) * FRAG_HALFS;
    for (int k = 0; k < 32; ++k) {
        float v = (k < d) ? (-2048.0f * row[k]) : ((k == d) ? 1024.0f * norm : 0.0f);
        const int lane = (k & 8) ? (nn + 16) : nn;
        const int h    = (k & 7) + ((k >= 16) ? 8 : 0);
        frag[lane * 16 + h] = (_Float16)v;
    }
}

// ---------------------------------------------------------------------------
// Main kernel: 256 threads (8 waves), 256 rows/block (2 row-tiles per wave).
// Per 16-code tile: one shared B fragment feeds TWO WMMAs (a0xB, a1xB); the
// argmin update of c0 co-executes in c1's WMMA hazard slots. B fragments are
// double-buffered in LDS and streamed with async global->LDS copies.
// ---------------------------------------------------------------------------
__global__ void __launch_bounds__(256) vq_main(
        const float* __restrict__ feat,
        const float* __restrict__ cb0, const float* __restrict__ cb1,
        const float* __restrict__ cb2, const float* __restrict__ cb3,
        const float* __restrict__ cb4,
        const _Float16* __restrict__ bpack,
        float* __restrict__ lossAcc,
        float* __restrict__ qout, int* __restrict__ idxOut) {

    __shared__ float ldsF[ROWS_PER_BLOCK * FDIM];                    // 36 KB
    __shared__ __align__(32) _Float16 ldsB[2][CHUNK_FRAG_HALFS];     // 2 x 32 KB

    const int tid  = threadIdx.x;
    const int lane = tid & 31;
    const int wave = tid >> 5;
    const int blockRow = blockIdx.x * ROWS_PER_BLOCK;

    for (int i = tid; i < ROWS_PER_BLOCK * FDIM; i += 256)
        ldsF[i] = feat[(size_t)blockRow * FDIM + i];

    const float* cbs[NSLICE] = {cb0, cb1, cb2, cb3, cb4};
    constexpr int OFF[NSLICE] = {0, 10, 16, 22, 31};
    constexpr int DIM[NSLICE] = {10, 6, 6, 9, 5};

    const int half    = lane >> 4;
    const int M       = lane & 15;
    const int waveRow = wave * 32;

#define PREFETCH_CHUNK(bufIdx, cidx)                                          \
    do {                                                                      \
        char* _src = (char*)bpack + (size_t)(cidx) * CHUNK_BYTES;             \
        char* _dst = (char*)&ldsB[bufIdx][0];                                 \
        _Pragma("unroll")                                                     \
        for (int _j = 0; _j < B128_PER_THREAD; ++_j) {                        \
            const int _i = (_j * 256 + tid) * 16;                             \
            ASYNC_COPY16(_src + _i, _dst + _i);                               \
        }                                                                     \
    } while (0)

    PREFETCH_CHUNK(0, 0);
    __syncthreads();        // also publishes ldsF

#pragma unroll
    for (int s = 0; s < NSLICE; ++s) {
        const int d = DIM[s], off = OFF[s];

        // Two A fragments (rows waveRow..+15 and +16..+31), 16x32 f16 layout:
        // lane<16 holds K {0..7,16..23}, lane>=16 holds K {8..15,24..31};
        // k==d is the constant-1 column multiplying the packed norm row.
        v16h a0, a1;
#pragma unroll
        for (int h = 0; h < 16; ++h) {
            const int k = ((h < 8) ? h : h + 8) + half * 8;
            float v0, v1;
            if (k < d) {
                v0 = ldsF[(waveRow + M) * FDIM + off + k];
                v1 = ldsF[(waveRow + 16 + M) * FDIM + off + k];
            } else {
                v0 = v1 = (k == d) ? 1.0f : 0.0f;
            }
            a0[h] = (_Float16)v0;
            a1[h] = (_Float16)v1;
        }

        float mval[2][8];
        int   midx[2][8];
#pragma unroll
        for (int g = 0; g < 2; ++g)
#pragma unroll
            for (int r = 0; r < 8; ++r) { mval[g][r] = 3.0e38f; midx[g][r] = 0; }

#pragma unroll
        for (int chunk = 0; chunk < NCHUNK; ++chunk) {
            const int cidx = s * NCHUNK + chunk;
            const int cur  = cidx & 1;

            __syncthreads();                 // other buffer free to overwrite
            if (cidx + 1 < NSLICE * NCHUNK) {
                PREFETCH_CHUNK(cur ^ 1, cidx + 1);
                ASYNC_WAIT(B128_PER_THREAD); // current chunk's loads have landed
            } else {
                ASYNC_WAIT(0);
            }
            __syncthreads();                 // all waves' copies visible

            const v16h* bv = (const v16h*)&ldsB[cur][0];
            for (int t = 0; t < TILES_PER_CHUNK; ++t) {
                v16h b = bv[t * 32 + lane];
                v8f c0 = {};
                v8f c1 = {};
                c0 = __builtin_amdgcn_wmma_f32_16x16x32_f16(
                         false, a0, false, b, (short)0, c0, false, false);
                c1 = __builtin_amdgcn_wmma_f32_16x16x32_f16(
                         false, a1, false, b, (short)0, c1, false, false);
                const int n = (chunk * TILES_PER_CHUNK + t) * 16 + M;
#pragma unroll
                for (int r = 0; r < 8; ++r) {
                    if (c0[r] < mval[0][r]) { mval[0][r] = c0[r]; midx[0][r] = n; }
                    if (c1[r] < mval[1][r]) { mval[1][r] = c1[r]; midx[1][r] = n; }
                }
            }
        }

        // Butterfly min-reduce across the 16 lanes of each half (wave32).
#pragma unroll
        for (int g = 0; g < 2; ++g) {
#pragma unroll
            for (int msk = 1; msk < 16; msk <<= 1) {
#pragma unroll
                for (int r = 0; r < 8; ++r) {
                    float ov = __shfl_xor(mval[g][r], msk, 32);
                    int   oi = __shfl_xor(midx[g][r], msk, 32);
                    if (ov < mval[g][r] || (ov == mval[g][r] && oi < midx[g][r])) {
                        mval[g][r] = ov; midx[g][r] = oi;
                    }
                }
            }
            float lsum = 0.0f;
            if (M == 0) {    // lanes 0 / 16 own rows (g*16 + 0..7) / (g*16 + 8..15)
                const int mbase = half * 8;
#pragma unroll
                for (int r = 0; r < 8; ++r) {
                    const int lrow = waveRow + g * 16 + mbase + r;
                    const int grow = blockRow + lrow;
                    const int n = midx[g][r];
                    idxOut[grow * NSLICE + s] = n;
                    const float* qrow = cbs[s] + n * d;
                    for (int k = 0; k < d; ++k) {
                        const float qv = qrow[k];
                        const float fv = ldsF[lrow * FDIM + off + k];
                        qout[(size_t)grow * FDIM + off + k] = qv;
                        const float df = qv - fv;
                        lsum += df * df;
                    }
                }
            }
            lsum += __shfl_xor(lsum, 16, 32);
            if (lane == 0) atomicAdd(&lossAcc[s], lsum);
        }
    }
#undef PREFETCH_CHUNK
}

// ---------------------------------------------------------------------------
// Epilogue: vq_loss = mean_s( 1.25 * sumsq_s / (B * d_s) )
// ---------------------------------------------------------------------------
__global__ void vq_finish(const float* __restrict__ lossAcc, float* __restrict__ out) {
    if (threadIdx.x == 0 && blockIdx.x == 0) {
        const int DIM[NSLICE] = {10, 6, 6, 9, 5};
        float acc = 0.0f;
        for (int s = 0; s < NSLICE; ++s)
            acc += 1.25f * lossAcc[s] / (float)((long long)B_TOTAL * DIM[s]);
        *out = acc * (1.0f / NSLICE);
    }
}

extern "C" void kernel_launch(void* const* d_in, const int* in_sizes, int n_in,
                              void* d_out, int out_size, void* d_ws, size_t ws_size,
                              hipStream_t stream) {
    const float* feat = (const float*)d_in[0];
    const float* cb0  = (const float*)d_in[1];
    const float* cb1  = (const float*)d_in[2];
    const float* cb2  = (const float*)d_in[3];
    const float* cb3  = (const float*)d_in[4];
    const float* cb4  = (const float*)d_in[5];

    _Float16* bpack  = (_Float16*)d_ws;
    float* lossAcc   = (float*)((char*)d_ws +
                        (size_t)NSLICE * SLICE_FRAG_HALFS * sizeof(_Float16));

    float* qout   = (float*)d_out;                     // [B,36]
    float* vqLoss = qout + (size_t)B_TOTAL * FDIM;     // scalar
    int*   idxOut = (int*)(vqLoss + 1);                // [B,5] int32 bits

    vq_pack<<<(NSLICE * KCODES + 255) / 256, 256, 0, stream>>>(
        cb0, cb1, cb2, cb3, cb4, bpack, lossAcc);

    vq_main<<<B_TOTAL / ROWS_PER_BLOCK, 256, 0, stream>>>(
        feat, cb0, cb1, cb2, cb3, cb4, bpack, lossAcc, qout, idxOut);

    vq_finish<<<1, 32, 0, stream>>>(lossAcc, vqLoss);
}
